// LongMaster_48189533061198
// MI455X (gfx1250) — compile-verified, matrix operand order checked by reading, not run on hardware
//
#include <hip/hip_runtime.h>

// LSTM: T=512, B=256, INPUT=256, HID=64, 4H=256, OUT=1024
#define TT 512
#define BB 256
#define IN 256
#define GG 256   // 4*HID
#define HH 64
#define OO 1024

typedef __attribute__((ext_vector_type(16))) __bf16 v16bf;
typedef __attribute__((ext_vector_type(8)))  __bf16 v8bf;
typedef __attribute__((ext_vector_type(8)))  float  v8f;

union Frag {
    v16bf v;
    v8bf  h[2];
    __bf16 e[16];
};

static __device__ __forceinline__ v8f wmma_bf16(v16bf a, v16bf b, v8f c) {
    return __builtin_amdgcn_wmma_f32_16x16x32_bf16(
        /*neg_a=*/false, a, /*neg_b=*/false, b,
        /*c_mod=*/(short)0, c, /*reuse_a=*/false, /*reuse_b=*/false);
}

// Build a 16x32 bf16 A-fragment chunk from fp32 global memory.
// p points at row base + (kc*32 + (lane>>4)*8); lane l holds row (l&15),
// K = base+{0..7} and base+{16..23} per ISA VGPR layout.
static __device__ __forceinline__ v16bf load_a_f32(const float* __restrict__ p) {
    Frag f;
#pragma unroll
    for (int q = 0; q < 8; ++q) {
        f.e[q]     = (__bf16)p[q];
        f.e[8 + q] = (__bf16)p[16 + q];
    }
    return f.v;
}

static __device__ __forceinline__ v16bf load_frag_bf16(const __bf16* __restrict__ p) {
    Frag f;
    f.h[0] = *(const v8bf*)p;
    f.h[1] = *(const v8bf*)(p + 16);
    return f.v;
}

static __device__ __forceinline__ float sigmoidf(float x) {
    return 1.0f / (1.0f + __expf(-x));
}

// ---------------------------------------------------------------------------
// Kernel 1: xg[m][g] = sum_k x[m][k]*W_ih[g][k] + (b_ih[g]+b_hh[g])
// M = T*B = 131072, N = 256, K = 256. Whole W_ih lives in LDS as bf16 (128 KB
// of the WGP's 320 KB). Block tile: 64(M) x 256(N); 8 waves, each wave does
// one 16-row m-tile and 8 n-tiles; K in 8 chunks of 32. A-fragments are
// double-buffered across K-chunks and B-fragments pipelined one tile ahead
// so ds/global latency overlaps the WMMA pipe.
// ---------------------------------------------------------------------------
__global__ __launch_bounds__(256) void k_xgates(const float* __restrict__ x,
                                                const float* __restrict__ Wih,
                                                const float* __restrict__ bih,
                                                const float* __restrict__ bhh,
                                                float* __restrict__ xg) {
    __shared__ __align__(16) __bf16 sW[GG * IN];  // 128 KB, [n][k] row-major
    const int tid = threadIdx.x;
    for (int i = tid; i < GG * IN; i += 256) sW[i] = (__bf16)Wih[i];
    __syncthreads();

    const int wave = tid >> 5, lane = tid & 31;
    const int lrow = lane & 15, lhi = lane >> 4;
    const int mrow0 = blockIdx.x * 64 + (wave & 3) * 16;
    const int nbase = (wave >> 2) * 128;  // two wave-groups split N=256

    v8f acc[8];
#pragma unroll
    for (int j = 0; j < 8; ++j)
        acc[j] = (v8f){0.f, 0.f, 0.f, 0.f, 0.f, 0.f, 0.f, 0.f};

    const float* xrow = x + (size_t)(mrow0 + lrow) * IN + lhi * 8;
    const __bf16* wbase = sW + (size_t)lrow * IN + lhi * 8;

    v16bf a = load_a_f32(xrow);  // kc = 0
#pragma unroll
    for (int kc = 0; kc < 8; ++kc) {
        // prefetch next A chunk from global while this chunk's WMMAs run
        v16bf anxt = a;
        if (kc + 1 < 8) anxt = load_a_f32(xrow + (kc + 1) * 32);

        // pipeline B fragments one n-tile ahead
        v16bf bcur = load_frag_bf16(wbase + (size_t)nbase * IN + kc * 32);
#pragma unroll
        for (int j = 0; j < 8; ++j) {
            v16bf bnxt = bcur;
            if (j + 1 < 8)
                bnxt = load_frag_bf16(wbase + (size_t)(nbase + (j + 1) * 16) * IN + kc * 32);
            acc[j] = wmma_bf16(a, bcur, acc[j]);
            bcur = bnxt;
        }
        a = anxt;
    }

#pragma unroll
    for (int j = 0; j < 8; ++j) {
        const int n = nbase + j * 16 + lrow;
        const float bias = bih[n] + bhh[n];
#pragma unroll
        for (int r = 0; r < 8; ++r) {
            const int m = mrow0 + r + lhi * 8;
            xg[(size_t)m * GG + n] = acc[j][r] + bias;
        }
    }
}

// ---------------------------------------------------------------------------
// Kernel 2: persistent recurrent scan. Batch rows are independent, so each of
// the 16 blocks owns 16 batch rows and runs all T=512 steps with W_hh (bf16,
// 32 KB) and h resident in LDS. Per step each of the 8 waves computes two
// 16x16 gate tiles (K=64 -> 2 bf16 WMMAs each), gates staged in LDS fp32,
// then elementwise LSTM cell update in fp32 registers. The next timestep's
// xg slice (16 KB/block) is prefetched (global_prefetch_b8) during the
// current step to hide HBM latency inside the serial dependence chain.
// Emits hs as bf16 for kernel 3's A-fragments.
// ---------------------------------------------------------------------------
__global__ __launch_bounds__(256) void k_lstm(const float* __restrict__ xg,
                                              const float* __restrict__ h0,
                                              const float* __restrict__ c0,
                                              const float* __restrict__ Whh,
                                              __bf16* __restrict__ hs,
                                              float* __restrict__ hT,
                                              float* __restrict__ cT) {
    __shared__ __align__(16) __bf16 sW[GG * HH];  // 32 KB, [n][k]
    __shared__ __align__(16) __bf16 sH[16 * HH];  // 2 KB,  [m][k]
    __shared__ __align__(16) float  sG[16 * GG];  // 16 KB gate staging

    const int tid = threadIdx.x;
    const int b0 = blockIdx.x * 16;
    for (int i = tid; i < GG * HH; i += 256) sW[i] = (__bf16)Whh[i];

    // Each thread owns 4 consecutive (b, hid) cells: p = tid*4 + q.
    float hreg[4], creg[4];
#pragma unroll
    for (int q = 0; q < 4; ++q) {
        const int p = tid * 4 + q;
        const int b = p >> 6, hid = p & 63;
        const float hv = h0[(size_t)(b0 + b) * HH + hid];
        creg[q] = c0[(size_t)(b0 + b) * HH + hid];
        hreg[q] = hv;
        sH[b * HH + hid] = (__bf16)hv;
    }
    // prefetch the first timestep's xg slice
    __builtin_prefetch((const char*)(xg + (size_t)b0 * GG) + tid * 64, 0, 3);
    __syncthreads();

    const int wave = tid >> 5, lane = tid & 31;
    const int lrow = lane & 15, lhi = lane >> 4;

    for (int t = 0; t < TT; ++t) {
        // pull next step's xg slice toward the caches while we compute
        if (t + 1 < TT)
            __builtin_prefetch(
                (const char*)(xg + ((size_t)(t + 1) * BB + b0) * GG) + tid * 64, 0, 3);

        // gates(16x256) = xg_t + h(16x64) @ W_hh^T ; wave w -> n-tiles 2w, 2w+1
#pragma unroll
        for (int jt = 0; jt < 2; ++jt) {
            const int n = (wave * 2 + jt) * 16 + lrow;
            v8f acc;
            const float* xgb = xg + ((size_t)t * BB + b0) * GG + n;
#pragma unroll
            for (int r = 0; r < 8; ++r)
                acc[r] = xgb[(size_t)(r + lhi * 8) * GG];
#pragma unroll
            for (int kc = 0; kc < 2; ++kc) {
                v16bf a = load_frag_bf16(sH + lrow * HH + kc * 32 + lhi * 8);
                v16bf b = load_frag_bf16(sW + n * HH + kc * 32 + lhi * 8);
                acc = wmma_bf16(a, b, acc);
            }
#pragma unroll
            for (int r = 0; r < 8; ++r)
                sG[(r + lhi * 8) * GG + n] = acc[r];
        }
        __syncthreads();

        // elementwise cell update (PyTorch gate order i,f,g,o)
#pragma unroll
        for (int q = 0; q < 4; ++q) {
            const int p = tid * 4 + q;
            const int b = p >> 6, hid = p & 63;
            const float ig = sigmoidf(sG[b * GG + hid]);
            const float fg = sigmoidf(sG[b * GG + 64 + hid]);
            const float gg = tanhf(sG[b * GG + 128 + hid]);
            const float og = sigmoidf(sG[b * GG + 192 + hid]);
            const float cv = fg * creg[q] + ig * gg;
            const float hv = og * tanhf(cv);
            creg[q] = cv;
            hreg[q] = hv;
            sH[b * HH + hid] = (__bf16)hv;
            hs[((size_t)t * BB + b0 + b) * HH + hid] = (__bf16)hv;
        }
        __syncthreads();
    }

#pragma unroll
    for (int q = 0; q < 4; ++q) {
        const int p = tid * 4 + q;
        const int b = p >> 6, hid = p & 63;
        hT[(size_t)(b0 + b) * HH + hid] = hreg[q];
        cT[(size_t)(b0 + b) * HH + hid] = creg[q];
    }
}

// ---------------------------------------------------------------------------
// Kernel 3: out = sigmoid(hs @ W_out^T + b_out). M = 131072, N = 1024, K = 64.
// Full W_out in LDS as bf16 (128 KB). Block tile 64(M) x 1024(N); each wave
// keeps its A-fragments (bf16, straight from global hs via b128 loads) in
// registers and sweeps 32 n-tiles with B-fragments pipelined one tile ahead.
// HBM-bound on the 512 MB output stream.
// ---------------------------------------------------------------------------
__global__ __launch_bounds__(256) void k_outproj(const __bf16* __restrict__ hs,
                                                 const float* __restrict__ Wout,
                                                 const float* __restrict__ bout,
                                                 float* __restrict__ out) {
    __shared__ __align__(16) __bf16 sW[OO * HH];  // 128 KB, [n][k]
    const int tid = threadIdx.x;
    for (int i = tid; i < OO * HH; i += 256) sW[i] = (__bf16)Wout[i];
    __syncthreads();

    const int wave = tid >> 5, lane = tid & 31;
    const int lrow = lane & 15, lhi = lane >> 4;
    const int mrow0 = blockIdx.x * 64 + (wave & 3) * 16;
    const int ntbase = (wave >> 2) * 32;

    // A fragments: K=64 -> two 32-K chunks, reused across all 32 n-tiles.
    const __bf16* ap = hs + (size_t)(mrow0 + lrow) * HH + lhi * 8;
    Frag a0, a1;
    a0.h[0] = *(const v8bf*)(ap + 0);
    a0.h[1] = *(const v8bf*)(ap + 16);
    a1.h[0] = *(const v8bf*)(ap + 32);
    a1.h[1] = *(const v8bf*)(ap + 48);

    const __bf16* wbase = sW + (size_t)lrow * HH + lhi * 8;

    // pipeline B fragments one n-tile ahead of the WMMA + store work
    v16bf b0c = load_frag_bf16(wbase + (size_t)(ntbase * 16) * HH);
    v16bf b1c = load_frag_bf16(wbase + (size_t)(ntbase * 16) * HH + 32);
    for (int jt = 0; jt < 32; ++jt) {
        v16bf b0n = b0c, b1n = b1c;
        if (jt + 1 < 32) {
            const __bf16* bp = wbase + (size_t)((ntbase + jt + 1) * 16) * HH;
            b0n = load_frag_bf16(bp);
            b1n = load_frag_bf16(bp + 32);
        }
        v8f acc = (v8f){0.f, 0.f, 0.f, 0.f, 0.f, 0.f, 0.f, 0.f};
        acc = wmma_bf16(a0.v, b0c, acc);
        acc = wmma_bf16(a1.v, b1c, acc);

        const int n = (ntbase + jt) * 16 + lrow;
        const float bias = bout[n];
#pragma unroll
        for (int r = 0; r < 8; ++r) {
            const float v = sigmoidf(acc[r] + bias);
            out[(size_t)(mrow0 + r + lhi * 8) * OO + n] = v;
        }
        b0c = b0n;
        b1c = b1n;
    }
}

extern "C" void kernel_launch(void* const* d_in, const int* in_sizes, int n_in,
                              void* d_out, int out_size, void* d_ws, size_t ws_size,
                              hipStream_t stream) {
    const float* x    = (const float*)d_in[0];
    const float* h0   = (const float*)d_in[1];
    const float* c0   = (const float*)d_in[2];
    const float* W_ih = (const float*)d_in[3];
    const float* W_hh = (const float*)d_in[4];
    const float* b_ih = (const float*)d_in[5];
    const float* b_hh = (const float*)d_in[6];
    const float* W_out = (const float*)d_in[7];
    const float* b_out = (const float*)d_in[8];

    float* out = (float*)d_out;
    float* hT  = out + (size_t)TT * BB * OO;   // outputs concatenated flat
    float* cT  = hT + (size_t)BB * HH;

    // workspace: xg fp32 (128 MB) then hs bf16 (16 MB)
    float*  xg = (float*)d_ws;
    __bf16* hs = (__bf16*)((char*)d_ws + (size_t)TT * BB * GG * sizeof(float));

    k_xgates<<<(TT * BB) / 64, 256, 0, stream>>>(x, W_ih, b_ih, b_hh, xg);
    k_lstm<<<BB / 16, 256, 0, stream>>>(xg, h0, c0, W_hh, hs, hT, cT);
    k_outproj<<<(TT * BB) / 64, 256, 0, stream>>>(hs, W_out, b_out, out);
}
